// GraphImputer_89593017794875
// MI455X (gfx1250) — matrix operand order repaired and checked by reading, not compiled, not run on hardware
//
#include <hip/hip_runtime.h>
#include <math.h>

// ---------------- problem constants (from reference) ----------------
#define N_NODES 22            // N = 2*NP_
#define NFEAT   6             // NF
#define HID     256
#define VARD    64
#define BSZ     128
#define T_STEPS 512
#define FEAT    (N_NODES*NFEAT)   // 132
#define MROWS   (BSZ*N_NODES)     // 2816  (== 22 * 128, multiple of TM)
#define EPSF    1.1920929e-07f

typedef __attribute__((ext_vector_type(16))) __bf16 v16bf;
typedef __attribute__((ext_vector_type(8)))  __bf16 v8bf;
typedef __attribute__((ext_vector_type(8)))  float  v8f;

// ---------------- WMMA GEMM:  C[M,N] = act( A[M,K] @ W[N,K]^T + bias (+C) ) ----
// Contract (enforced by host-side padding):
//   M % 128 == 0, K % 32 == 0, lda % 4 == 0, ldw % 4 == 0, 16B-aligned A/W.
// Block tile 128x64, 256 threads = 8 waves, each wave a 32x32 patch (2x2 wmma).
// Staging: per-thread contiguous float4 global loads -> bf16 -> 16B LDS stores,
// register double-buffered across K chunks.
#define TM 128
#define TN 64
#define KC 32
#define KCP 40   // LDS row stride in bf16 (80B: 16B-aligned, conflict-free)

template<int ACT, bool ACCUM>
__global__ __launch_bounds__(256)
void wmma_gemm(const float* __restrict__ A, int lda,
               const float* __restrict__ W, int ldw,     // W is [Nc x K] row-major
               const float* __restrict__ bias,           // nullable
               float* __restrict__ C, int ldc,
               int Nc, int K)
{
    __shared__ __align__(16) __bf16 As[TM][KCP];
    __shared__ __align__(16) __bf16 Bs[TN][KCP];

    const int tid  = threadIdx.x;
    const int lane = tid & 31;
    const int wid  = tid >> 5;
    const int wm   = (wid & 3) * 32;   // wave row offset inside block tile
    const int wn   = (wid >> 2) * 32;  // wave col offset inside block tile
    const int m0   = blockIdx.y * TM;
    const int n0   = blockIdx.x * TN;
    const int half = lane >> 4;        // 0: lanes 0-15, 1: lanes 16-31
    const int l    = lane & 15;

    // staging ownership: A -> 16 contiguous floats/thread, B -> 8 contiguous
    const int ar = tid >> 1;           // 0..127
    const int ac = (tid & 1) * 16;     // 0 or 16
    const int br = tid >> 2;           // 0..63
    const int bc = (tid & 3) * 8;      // 0,8,16,24
    const bool bok = (n0 + br) < Nc;   // uniform per thread, hoisted

    const float* __restrict__ Aptr = A + (size_t)(m0 + ar) * lda + ac;
    const float* __restrict__ Wptr = W + (size_t)(n0 + br) * ldw + bc;

    float areg[16];
    float breg[8];
    // prologue: load chunk k0 = 0
#pragma unroll
    for (int q = 0; q < 4; ++q) {
        float4 t = *(const float4*)(Aptr + 4 * q);
        areg[4*q+0] = t.x; areg[4*q+1] = t.y; areg[4*q+2] = t.z; areg[4*q+3] = t.w;
    }
    if (bok) {
        float4 t0 = *(const float4*)(Wptr);
        float4 t1 = *(const float4*)(Wptr + 4);
        breg[0]=t0.x; breg[1]=t0.y; breg[2]=t0.z; breg[3]=t0.w;
        breg[4]=t1.x; breg[5]=t1.y; breg[6]=t1.z; breg[7]=t1.w;
    } else {
#pragma unroll
        for (int j = 0; j < 8; ++j) breg[j] = 0.f;
    }

    v8f acc[2][2];
#pragma unroll
    for (int a = 0; a < 2; ++a)
#pragma unroll
        for (int b = 0; b < 2; ++b)
#pragma unroll
            for (int v = 0; v < 8; ++v) acc[a][b][v] = 0.f;

    for (int k0 = 0; k0 < K; k0 += KC) {
        // stage current regs -> LDS (fp32->bf16, 16B stores)
        {
            v8bf p0, p1, q0;
#pragma unroll
            for (int j = 0; j < 8; ++j) {
                p0[j] = (__bf16)areg[j];
                p1[j] = (__bf16)areg[8 + j];
                q0[j] = (__bf16)breg[j];
            }
            *(v8bf*)&As[ar][ac]     = p0;
            *(v8bf*)&As[ar][ac + 8] = p1;
            *(v8bf*)&Bs[br][bc]     = q0;
        }
        __syncthreads();

        // issue next chunk's global loads early (overlap with MMA below)
        if (k0 + KC < K) {
            const int kn = k0 + KC;
#pragma unroll
            for (int q = 0; q < 4; ++q) {
                float4 t = *(const float4*)(Aptr + kn + 4 * q);
                areg[4*q+0] = t.x; areg[4*q+1] = t.y; areg[4*q+2] = t.z; areg[4*q+3] = t.w;
            }
            if (bok) {
                float4 t0 = *(const float4*)(Wptr + kn);
                float4 t1 = *(const float4*)(Wptr + kn + 4);
                breg[0]=t0.x; breg[1]=t0.y; breg[2]=t0.z; breg[3]=t0.w;
                breg[4]=t1.x; breg[5]=t1.y; breg[6]=t1.z; breg[7]=t1.w;
                if (k0 + 2 * KC < K) __builtin_prefetch(Wptr + k0 + 2 * KC, 0, 1);
            }
        }

        // B fragments: 32x16, lane=column, lanes0-15 K 0..15 / lanes16-31 K 16..31
        v16bf bfr[2];
#pragma unroll
        for (int nt = 0; nt < 2; ++nt) {
            int col = wn + nt * 16 + l;
            v8bf lo = *(v8bf*)&Bs[col][half * 16];
            v8bf hi = *(v8bf*)&Bs[col][half * 16 + 8];
#pragma unroll
            for (int j = 0; j < 8; ++j) { bfr[nt][j] = lo[j]; bfr[nt][8 + j] = hi[j]; }
        }
        // A fragments: 16x32, lane=row; lanes0-15 K 0..15, lanes16-31 K 16..31
#pragma unroll
        for (int mt = 0; mt < 2; ++mt) {
            int row = wm + mt * 16 + l;
            v8bf lo = *(v8bf*)&As[row][half * 16];
            v8bf hi = *(v8bf*)&As[row][half * 16 + 8];
            v16bf af;
#pragma unroll
            for (int j = 0; j < 8; ++j) { af[j] = lo[j]; af[8 + j] = hi[j]; }
#pragma unroll
            for (int nt = 0; nt < 2; ++nt)
                acc[mt][nt] = __builtin_amdgcn_wmma_f32_16x16x32_bf16(
                    false, af, false, bfr[nt], (short)0, acc[mt][nt], false, false);
        }
        __syncthreads();
    }

    // epilogue: lanes 0-15: M=v, N=lane ; lanes 16-31: M=8+v, N=lane-16. M unguarded.
#pragma unroll
    for (int nt = 0; nt < 2; ++nt) {
        int col = n0 + wn + nt * 16 + l;
        if (col >= Nc) continue;
        float bv = bias ? bias[col] : 0.f;
#pragma unroll
        for (int mt = 0; mt < 2; ++mt) {
#pragma unroll
            for (int v = 0; v < 8; ++v) {
                int row = m0 + wm + mt * 16 + half * 8 + v;
                float r = acc[mt][nt][v] + bv;
                if (ACCUM) r += C[(size_t)row * ldc + col];
                if (ACT == 1) r = r > 0.f ? r : 0.f;                    // relu
                if (ACT == 2) r = (r > 20.f) ? r : log1pf(expf(r));     // softplus
                C[(size_t)row * ldc + col] = r;
            }
        }
    }
}

// ---------------- pointwise / helper kernels ----------------
__device__ __forceinline__ float sigm(float x) { return 1.f / (1.f + expf(-x)); }

__global__ void zero_f32(float* __restrict__ p, long long n) {
    long long i = (long long)blockIdx.x * 256 + threadIdx.x;
    if (i < n) p[i] = 0.f;
}

__global__ void copy_prev(const float* __restrict__ x, float* __restrict__ prev) {
    int idx = blockIdx.x * 256 + threadIdx.x;
    if (idx >= BSZ * FEAT) return;
    int b = idx / FEAT, f = idx - b * FEAT;
    prev[idx] = x[(size_t)b * T_STEPS * FEAT + f];   // x[:,0,:]
}

// xpad[M x 32]: cols 0..5 = prev (viewed [M,6]), rest zero  (K padding for LSTM1 input GEMM)
__global__ void pad_prev(const float* __restrict__ prev, float* __restrict__ xpad) {
    int idx = blockIdx.x * 256 + threadIdx.x;
    if (idx >= MROWS * 32) return;
    int m = idx >> 5, c = idx & 31;
    xpad[idx] = (c < NFEAT) ? prev[m * NFEAT + c] : 0.f;
}

// wpad[1024 x 32] from lstm1_Wih[1024 x 6], zero-padded
__global__ void pad_wih(const float* __restrict__ W, float* __restrict__ Wp) {
    int idx = blockIdx.x * 256 + threadIdx.x;
    if (idx >= 4 * HID * 32) return;
    int o = idx >> 5, c = idx & 31;
    Wp[idx] = (c < NFEAT) ? W[o * NFEAT + c] : 0.f;
}

__global__ void fold_bias(const float* __restrict__ b1, const float* __restrict__ b2,
                          float* __restrict__ o, int n) {
    int i = blockIdx.x * 256 + threadIdx.x;
    if (i < n) o[i] = b1[i] + b2[i];
}

// e = [x,x] @ W^T == x @ (W[:, :din] + W[:, din:])^T ; output stride dpad, zero-padded
__global__ void fold_e1(const float* __restrict__ W, float* __restrict__ Weff,
                        int din, int dpad) {
    int i = blockIdx.x * 256 + threadIdx.x;
    if (i >= HID * dpad) return;
    int o = i / dpad, k = i - o * dpad;
    Weff[i] = (k < din) ? (W[(size_t)o * (2 * din) + k] + W[(size_t)o * (2 * din) + din + k])
                        : 0.f;
}

// out[M x wout] = [A | B | zeros]
__global__ void concat2(const float* __restrict__ A, int wa,
                        const float* __restrict__ B, int wb,
                        float* __restrict__ out, int wout) {
    long long idx = (long long)blockIdx.x * 256 + threadIdx.x;
    if (idx >= (long long)MROWS * wout) return;
    int m = (int)(idx / wout), c = (int)(idx - (long long)m * wout);
    float v = 0.f;
    if (c < wa)           v = A[(size_t)m * wa + c];
    else if (c < wa + wb) v = B[(size_t)m * wb + (c - wa)];
    out[idx] = v;
}

__global__ void lstm_update(const float* __restrict__ g1, const float* __restrict__ g2,
                            float* __restrict__ h1, float* __restrict__ c1,
                            float* __restrict__ h2, float* __restrict__ c2) {
    int idx = blockIdx.x * 256 + threadIdx.x;
    if (idx >= MROWS * HID) return;
    int m = idx >> 8, j = idx & 255;
    size_t gb = (size_t)m * 4 * HID;
    {
        float i = sigm(g1[gb + j]), f = sigm(g1[gb + HID + j]);
        float g = tanhf(g1[gb + 2 * HID + j]), o = sigm(g1[gb + 3 * HID + j]);
        float c = f * c1[idx] + i * g;
        c1[idx] = c; h1[idx] = o * tanhf(c);
    }
    {
        float i = sigm(g2[gb + j]), f = sigm(g2[gb + HID + j]);
        float g = tanhf(g2[gb + 2 * HID + j]), o = sigm(g2[gb + 3 * HID + j]);
        float c = f * c2[idx] + i * g;
        c2[idx] = c; h2[idx] = o * tanhf(c);
    }
}

// agg = broadcast(sum over node dim)
__global__ void node_agg(const float* __restrict__ e, float* __restrict__ agg) {
    int b = blockIdx.x, h = threadIdx.x;   // 256 threads
    float s = 0.f;
#pragma unroll
    for (int n = 0; n < N_NODES; ++n) s += e[((size_t)(b * N_NODES + n)) * HID + h];
#pragma unroll
    for (int n = 0; n < N_NODES; ++n) agg[((size_t)(b * N_NODES + n)) * HID + h] = s;
}

__global__ void z_sample(const float* __restrict__ eps_t, const float* __restrict__ mu,
                         const float* __restrict__ sd, float* __restrict__ z) {
    int idx = blockIdx.x * 256 + threadIdx.x;
    if (idx < MROWS * VARD) z[idx] = eps_t[idx] * sd[idx] + mu[idx];
}

// replace = where(mask==0, out+prev, x_t); writes outputs, updates prev,
// accumulates sum((out - input_diff)^2) into recons_acc.
__global__ void replace_out(const float* __restrict__ outb, const float* __restrict__ x,
                            const int* __restrict__ mask, float* __restrict__ prev,
                            float* __restrict__ dout, float* __restrict__ recons_acc, int t) {
    int idx = blockIdx.x * 256 + threadIdx.x;
    float local = 0.f;
    if (idx < BSZ * FEAT) {
        int b = idx / FEAT, f = idx - b * FEAT;
        int n = f / NFEAT, j = f - n * NFEAT;
        float ov = outb[((size_t)(b * N_NODES + n)) * 16 + j];
        size_t xoff = (size_t)b * T_STEPS * FEAT + (size_t)t * FEAT + f;
        float pv = prev[idx];
        float xt = x[xoff];
        float rep = (mask[xoff] == 0) ? (ov + pv) : xt;
        dout[xoff] = rep;
        prev[idx] = rep;
        float indiff = (t > 0) ? (xt - x[xoff - FEAT]) : 0.f;
        float d = ov - indiff;
        local = d * d;
    }
    __shared__ float red[256];
    red[threadIdx.x] = local; __syncthreads();
    for (int s = 128; s > 0; s >>= 1) {
        if (threadIdx.x < s) red[threadIdx.x] += red[threadIdx.x + s];
        __syncthreads();
    }
    if (threadIdx.x == 0) atomicAdd(recons_acc, red[0]);
}

__global__ void kld_reduce(const float* __restrict__ em, const float* __restrict__ es,
                           const float* __restrict__ pm, const float* __restrict__ ps,
                           float* __restrict__ kld_acc) {
    int idx = blockIdx.x * 256 + threadIdx.x;
    float local = 0.f;
    if (idx < MROWS * VARD) {
        float e = es[idx], p = ps[idx], dm = em[idx] - pm[idx];
        local = 2.f * logf(p + EPSF) - 2.f * logf(e + EPSF) + (e * e + dm * dm) / (p * p) - 1.f;
    }
    __shared__ float red[256];
    red[threadIdx.x] = local; __syncthreads();
    for (int s = 128; s > 0; s >>= 1) {
        if (threadIdx.x < s) red[threadIdx.x] += red[threadIdx.x + s];
        __syncthreads();
    }
    if (threadIdx.x == 0) atomicAdd(kld_acc, red[0]);
}

__global__ void finalize_loss(const float* __restrict__ scal, float* __restrict__ dout) {
    if (threadIdx.x == 0) {
        float recons = scal[0] / (float)((size_t)BSZ * T_STEPS * FEAT);
        float kld = 0.5f * scal[1];
        dout[(size_t)BSZ * T_STEPS * FEAT] = recons + 0.001f * kld;
    }
}

// ---------------- host orchestration ----------------
extern "C" void kernel_launch(void* const* d_in, const int* in_sizes, int n_in,
                              void* d_out, int out_size, void* d_ws, size_t ws_size,
                              hipStream_t stream) {
    (void)in_sizes; (void)out_size; (void)ws_size;
    const float* x    = (const float*)d_in[0];
    /* d_in[1] = masked_x: unused by the reference forward */
    const int*   mask = (const int*)d_in[2];
    const float* eps  = (const float*)d_in[3];
    const float* P[48];
    for (int i = 4; i < n_in && i < 52; ++i) P[i - 4] = (const float*)d_in[i];
    float* out = (float*)d_out;

    enum { LSTM1_WIH = 0, LSTM1_WHH, LSTM1_BIH, LSTM1_BHH,
           LSTM2_WIH, LSTM2_WHH, LSTM2_BIH, LSTM2_BHH,
           ENC_E1_W, ENC_E1_B, ENC_E2_W, ENC_E2_B, ENC_N1_W, ENC_N1_B, ENC_N2_W, ENC_N2_B,
           PRI_E1_W, PRI_E1_B, PRI_E2_W, PRI_E2_B, PRI_N1_W, PRI_N1_B, PRI_N2_W, PRI_N2_B,
           DEC_E1_W, DEC_E1_B, DEC_E2_W, DEC_E2_B, DEC_N1_W, DEC_N1_B, DEC_N2_W, DEC_N2_B,
           ENC_MEAN_W, ENC_MEAN_B, ENC_STD_W, ENC_STD_B,
           PRI_MEAN_W, PRI_MEAN_B, PRI_STD_W, PRI_STD_B,
           PHI_W, PHI_B, FINAL_W, FINAL_B };

    const int ENC_KP = 288;               // 262 padded to multiple of 32
    const int DEC_K  = VARD + HID;        // 320, already multiple of 32

    // workspace carve-out (all sizes multiples of 4 floats -> 16B alignment kept)
    float* base = (float*)d_ws; size_t off = 0;
    auto alloc = [&](size_t n) { float* r = base + off; off += n; return r; };
    const size_t MH = (size_t)MROWS * HID;
    float* h1 = alloc(MH); float* c1 = alloc(MH); float* h2 = alloc(MH); float* c2 = alloc(MH);
    float* g1 = alloc(MH * 4); float* g2 = alloc(MH * 4);
    float* prev = alloc((size_t)BSZ * FEAT);
    float* xpad = alloc((size_t)MROWS * 32);
    float* cat  = alloc((size_t)MROWS * DEC_K);          // max width 320 (>=288)
    float* ebuf = alloc(MH); float* e2buf = alloc(MH); float* aggbuf = alloc(MH);
    float* nbuf = alloc(MH); float* gnout = alloc(MH);
    float* mu_e = alloc((size_t)MROWS * VARD); float* sd_e = alloc((size_t)MROWS * VARD);
    float* mu_p = alloc((size_t)MROWS * VARD); float* sd_p = alloc((size_t)MROWS * VARD);
    float* zb = alloc((size_t)MROWS * VARD);   float* zf  = alloc((size_t)MROWS * VARD);
    float* outb = alloc((size_t)MROWS * 16);
    float* wihpad   = alloc((size_t)4 * HID * 32);       // 1024x32
    float* weff_enc = alloc((size_t)HID * ENC_KP);       // 256x288 (padded)
    float* weff_pri = alloc((size_t)HID * HID);          // 256x256
    float* weff_dec = alloc((size_t)HID * DEC_K);        // 256x320
    float* bsum1 = alloc(4 * HID); float* bsum2 = alloc(4 * HID);
    float* scal = alloc(4);   // [0]=recons sum, [1]=kld sum

    auto gemm = [&](int act, bool accum, const float* A, int lda,
                    const float* W, int ldw, const float* bias,
                    float* C, int ldc, int Nc, int K) {
        dim3 g((Nc + TN - 1) / TN, MROWS / TM);
        if (accum)            wmma_gemm<0, true ><<<g, 256, 0, stream>>>(A, lda, W, ldw, bias, C, ldc, Nc, K);
        else if (act == 1)    wmma_gemm<1, false><<<g, 256, 0, stream>>>(A, lda, W, ldw, bias, C, ldc, Nc, K);
        else if (act == 2)    wmma_gemm<2, false><<<g, 256, 0, stream>>>(A, lda, W, ldw, bias, C, ldc, Nc, K);
        else                  wmma_gemm<0, false><<<g, 256, 0, stream>>>(A, lda, W, ldw, bias, C, ldc, Nc, K);
    };

    // graphnet chain: e1(+relu, folded [x,x] weights) -> e2 -> node-sum agg -> n1(+relu) -> n2
    auto gnet = [&](const float* inA, int K, const float* weff, int pb, float* ob) {
        gemm(1, false, inA, K,    weff,       K,   P[pb + 1], ebuf,  HID, HID, K);
        gemm(0, false, ebuf, HID, P[pb + 2],  HID, P[pb + 3], e2buf, HID, HID, HID);
        node_agg<<<BSZ, HID, 0, stream>>>(e2buf, aggbuf);
        gemm(1, false, aggbuf, HID, P[pb + 4], HID, P[pb + 5], nbuf,  HID, HID, HID);
        gemm(0, false, nbuf, HID,  P[pb + 6],  HID, P[pb + 7], ob,    HID, HID, HID);
    };

    // ---- prep (fresh every call: deterministic, no cached state) ----
    zero_f32<<<(int)((4 * MH + 255) / 256), 256, 0, stream>>>(h1, (long long)(4 * MH)); // h1,c1,h2,c2 contiguous
    zero_f32<<<1, 256, 0, stream>>>(scal, 4);
    copy_prev<<<(BSZ * FEAT + 255) / 256, 256, 0, stream>>>(x, prev);
    pad_wih<<<(4 * HID * 32 + 255) / 256, 256, 0, stream>>>(P[LSTM1_WIH], wihpad);
    fold_bias<<<(4 * HID + 255) / 256, 256, 0, stream>>>(P[LSTM1_BIH], P[LSTM1_BHH], bsum1, 4 * HID);
    fold_bias<<<(4 * HID + 255) / 256, 256, 0, stream>>>(P[LSTM2_BIH], P[LSTM2_BHH], bsum2, 4 * HID);
    fold_e1<<<(HID * ENC_KP + 255) / 256, 256, 0, stream>>>(P[ENC_E1_W], weff_enc, NFEAT + HID, ENC_KP);
    fold_e1<<<(HID * HID + 255) / 256, 256, 0, stream>>>(P[PRI_E1_W], weff_pri, HID, HID);
    fold_e1<<<(HID * DEC_K + 255) / 256, 256, 0, stream>>>(P[DEC_E1_W], weff_dec, DEC_K, DEC_K);

    const int ptw = (MROWS * HID + 255) / 256;
    const int vtw = (MROWS * VARD + 255) / 256;

    // ---- sequential scan over time ----
    for (int t = 0; t < T_STEPS; ++t) {
        // LSTM gate GEMMs (padded input pass + recurrent accumulate pass)
        pad_prev<<<(MROWS * 32 + 255) / 256, 256, 0, stream>>>(prev, xpad);
        gemm(0, false, xpad, 32,  wihpad,       32,  bsum1,   g1, 4 * HID, 4 * HID, 32);
        gemm(0, true,  h1,   HID, P[LSTM1_WHH], HID, nullptr, g1, 4 * HID, 4 * HID, HID);
        gemm(0, false, h1,   HID, P[LSTM2_WIH], HID, bsum2,   g2, 4 * HID, 4 * HID, HID);
        gemm(0, true,  h2,   HID, P[LSTM2_WHH], HID, nullptr, g2, 4 * HID, 4 * HID, HID);
        lstm_update<<<ptw, 256, 0, stream>>>(g1, g2, h1, c1, h2, c2);   // h2 becomes "node"

        // encoder graphnet on [prev_rs, node] (padded to K=288)
        concat2<<<(int)(((long long)MROWS * ENC_KP + 255) / 256), 256, 0, stream>>>(
            prev, NFEAT, h2, HID, cat, ENC_KP);
        gnet(cat, ENC_KP, weff_enc, ENC_E1_W, gnout);
        gemm(0, false, gnout, HID, P[ENC_MEAN_W], HID, P[ENC_MEAN_B], mu_e, VARD, VARD, HID);
        gemm(2, false, gnout, HID, P[ENC_STD_W],  HID, P[ENC_STD_B],  sd_e, VARD, VARD, HID);

        // prior graphnet on node
        gnet(h2, HID, weff_pri, PRI_E1_W, gnout);
        gemm(0, false, gnout, HID, P[PRI_MEAN_W], HID, P[PRI_MEAN_B], mu_p, VARD, VARD, HID);
        gemm(2, false, gnout, HID, P[PRI_STD_W],  HID, P[PRI_STD_B],  sd_p, VARD, VARD, HID);

        // reparameterize + phi(+relu)
        z_sample<<<vtw, 256, 0, stream>>>(eps + (size_t)t * MROWS * VARD, mu_e, sd_e, zb);
        gemm(1, false, zb, VARD, P[PHI_W], VARD, P[PHI_B], zf, VARD, VARD, VARD);

        // decoder graphnet on [zf, node] (K=320) + final projection
        concat2<<<(int)(((long long)MROWS * DEC_K + 255) / 256), 256, 0, stream>>>(
            zf, VARD, h2, HID, cat, DEC_K);
        gnet(cat, DEC_K, weff_dec, DEC_E1_W, gnout);
        gemm(0, false, gnout, HID, P[FINAL_W], HID, P[FINAL_B], outb, 16, NFEAT, HID);

        // outputs / mask replace / running recon loss; updates prev
        replace_out<<<(BSZ * FEAT + 255) / 256, 256, 0, stream>>>(outb, x, mask, prev, out, scal, t);
    }

    // KLD from final-step stats + total loss scalar
    kld_reduce<<<vtw, 256, 0, stream>>>(mu_e, sd_e, mu_p, sd_p, scal + 1);
    finalize_loss<<<1, 64, 0, stream>>>(scal, out);
}
